// BLADEBlock_50981261803798
// MI455X (gfx1250) — matrix operand order repaired and checked by reading, not compiled, not run on hardware
//
#include <hip/hip_runtime.h>
#include <hip/hip_bf16.h>

typedef __attribute__((ext_vector_type(16))) _Float16 v16h;
typedef __attribute__((ext_vector_type(8)))  float    v8f;

#define D_     1024
#define H_     16
#define KVH_   4
#define DH_    64
#define CHUNK_ 256
#define SDIM_  128
#define MG_    8
#define B_     4
#define T_     4096
#define TC_    264          // MG + CHUNK (valid tokens)
#define TCP_   288          // padded Tc: multiple of 32 (K-dim) and 16 (tiles)
#define KD_    256          // KVH*DH
#define NCH_   16           // T / CHUNK
#define BH_    64           // B*H
#define SCALE_ 0.125f       // 1/sqrt(DH)
#define BTCP_  (B_ * TCP_)  // 1152 padded rows

#define LDSB_ROW_ 72        // padded LDS tile row stride (halves) -> bank-conflict free
#define LDSB_TILE_ (64 * LDSB_ROW_)

union FragU { v16h h; unsigned u[8]; };

__device__ __forceinline__ void async_copy_b128(unsigned lds_off, const void* gptr) {
  // GLOBAL_LOAD_ASYNC_TO_LDS_B128 (GV mode): vdst = LDS byte offset, vaddr = 64b addr
  asm volatile("global_load_async_to_lds_b128 %0, %1, off"
               :: "v"(lds_off), "v"(gptr) : "memory");
}

// =====================================================================
// WMMA GEMM core, LDS-staged B with async double buffering.
// Wave computes 32x64 of C[M,N] = A[M,K] * W^T (+bias)(+gelu),
// W row-major [N,K]. Block = 4 waves -> 128x64. Shapes compile-time.
// Per K-step (32): issue next 64x32 B tile via async-to-LDS, wait
// asynccnt, barrier, 2 A-frag global b128 loads + 8 ds-frag WMMAs.
// NG: clamp B row index & guard stores when N % 64 != 0 (scores only).
// =====================================================================
template <int M, int N, int K, int LDA, int LDB, int LDC,
          bool NG, int ACT, bool F16OUT, bool BIAS>
__device__ __forceinline__ void gemm_core(const _Float16* __restrict__ A,
                                          const _Float16* __restrict__ W,
                                          void* __restrict__ Cp,
                                          const float* __restrict__ bias) {
  __shared__ _Float16 ldsB[2][LDSB_TILE_];
  const int tid  = threadIdx.x;
  const int lane = tid & 31;
  const int l15  = lane & 15;
  const int wid  = tid >> 5;
  int row0 = blockIdx.y * 128 + wid * 32;
  if (row0 > M - 32) row0 = M - 32;            // clamp (M % 32 == 0): dup waves
  const int col0 = blockIdx.x * 64;

  // ---- async B-tile copy setup: thread t copies row t>>1, segments (t&1)*2,+1
  const int crow = tid >> 1;                   // 0..63: B row within tile
  const int seg0 = (tid & 1) * 2;              // b128 segment pair
  int bn = col0 + crow;
  if (NG) bn = (bn > N - 1) ? (N - 1) : bn;    // clamp: safe read, store guarded
  const _Float16* gB = W + (size_t)bn * LDB;
  const unsigned ldsBase = (unsigned)(uintptr_t)(&ldsB[0][0]) +
                           crow * (LDSB_ROW_ * 2) + seg0 * 16;

  const int ga = (lane & 16) ? 4 : 0;          // A dword-group offset
  const int gb = (lane & 16) ? 4 : 0;          // B dword-group offset
  const unsigned* Ar0 = (const unsigned*)(A + (size_t)(row0 + l15) * LDA);
  const unsigned* Ar1 = (const unsigned*)(A + (size_t)(row0 + 16 + l15) * LDA);

  // prime buffer 0 with K-chunk 0
  async_copy_b128(ldsBase, gB + seg0 * 8);
  async_copy_b128(ldsBase + 16, gB + seg0 * 8 + 8);

  v8f acc[2][4] = {};
  for (int k0 = 0; k0 < K; k0 += 32) {
    const int p = (k0 >> 5) & 1;
    if (k0 + 32 < K) {                         // issue next tile into other buffer
      const _Float16* g = gB + (k0 + 32) + seg0 * 8;
      const unsigned l = ldsBase + (p ^ 1) * (unsigned)(LDSB_TILE_ * 2);
      async_copy_b128(l, g);
      async_copy_b128(l + 16, g + 8);
      asm volatile("s_wait_asynccnt 0x2" ::: "memory");
    } else {
      asm volatile("s_wait_asynccnt 0x0" ::: "memory");
    }
    __syncthreads();                           // current tile visible to all waves

    const int kd = k0 >> 1;                    // dword offset within A row
    FragU a0, a1;
#pragma unroll
    for (int v = 0; v < 8; ++v) {
      const int ai = kd + (v & 3) + ((v >> 2) << 3) + ga;
      a0.u[v] = Ar0[ai];
      a1.u[v] = Ar1[ai];
    }
    if (k0 + 64 < K) {
      __builtin_prefetch((const void*)(Ar0 + kd + 32), 0, 1);
      __builtin_prefetch((const void*)(Ar1 + kd + 32), 0, 1);
    }
    const unsigned* bt = (const unsigned*)&ldsB[p][0];
#pragma unroll
    for (int j = 0; j < 4; ++j) {
      FragU bf;
      const int bbase = (16 * j + l15) * (LDSB_ROW_ / 2) + gb;
#pragma unroll
      for (int v = 0; v < 8; ++v) bf.u[v] = bt[bbase + (v & 3) + ((v >> 2) << 3)];
      acc[0][j] = __builtin_amdgcn_wmma_f32_16x16x32_f16(
          false, a0.h, false, bf.h, (short)0, acc[0][j], false, false);
      acc[1][j] = __builtin_amdgcn_wmma_f32_16x16x32_f16(
          false, a1.h, false, bf.h, (short)0, acc[1][j], false, false);
    }
    __syncthreads();                           // readers done before next overwrite
  }

  // C/D layout: VGPR r -> M = base+r (lanes 0-15) / base+8+r (lanes 16-31)
#pragma unroll
  for (int rt = 0; rt < 2; ++rt) {
    const int rbase = row0 + rt * 16 + ((lane & 16) ? 8 : 0);
#pragma unroll
    for (int j = 0; j < 4; ++j) {
      const int cn = col0 + 16 * j + l15;
      if (NG && cn >= N) continue;
      const float bv = BIAS ? bias[cn] : 0.f;
#pragma unroll
      for (int r = 0; r < 8; ++r) {
        const int cm = rbase + r;
        float val = acc[rt][j][r] + bv;
        if (ACT == 1) val = 0.5f * val * (1.f + erff(val * 0.70710678118654752f));
        if (F16OUT) ((_Float16*)Cp)[(size_t)cm * LDC + cn] = (_Float16)val;
        else        ((float*)Cp)[(size_t)cm * LDC + cn] = val;
      }
    }
  }
}

template <int M, int N, int K, int LDA, int LDB, int LDC,
          bool NG, int ACT, bool F16OUT, bool BIAS>
__global__ __launch_bounds__(128) void gemm_tpl(const _Float16* __restrict__ A,
                                                const _Float16* __restrict__ W,
                                                void* __restrict__ C,
                                                const float* __restrict__ bias) {
  gemm_core<M, N, K, LDA, LDB, LDC, NG, ACT, F16OUT, BIAS>(A, W, C, bias);
}

// scores[b,h] = Q_h * K_kvh^T : per (b,h) 288x288, K=64 (N clamped)
__global__ __launch_bounds__(128) void attn_scores_kernel(
    const _Float16* __restrict__ q, const _Float16* __restrict__ k,
    float* __restrict__ sc) {
  int z = blockIdx.z, b = z >> 4, qh = z & 15, kvh = qh >> 2;
  gemm_core<TCP_, TCP_, DH_, D_, KD_, TCP_, true, 0, false, false>(
      q + (size_t)b * TCP_ * D_ + qh * DH_,
      k + (size_t)b * TCP_ * KD_ + kvh * DH_,
      sc + (size_t)z * TCP_ * TCP_, nullptr);
}

// O[b,h] = attn[b,h] * V_kvh : 288x64, K=288, V pre-transposed to [KD][TCP]
__global__ __launch_bounds__(128) void attn_av_kernel(
    const _Float16* __restrict__ attn, const _Float16* __restrict__ vt,
    _Float16* __restrict__ o) {
  int z = blockIdx.z, b = z >> 4, qh = z & 15, kvh = qh >> 2;
  gemm_core<TCP_, DH_, TCP_, TCP_, TCP_, D_, false, 0, true, false>(
      attn + (size_t)z * TCP_ * TCP_,
      vt + ((size_t)b * KD_ + kvh * DH_) * TCP_,
      o + (size_t)b * TCP_ * D_ + qh * DH_, nullptr);
}

// ---------------- elementwise / reduction kernels ----------------
__global__ void f32_to_f16_kernel(const float* src, _Float16* dst, int n) {
  int i = blockIdx.x * 256 + threadIdx.x;
  if (i < n) dst[i] = (_Float16)src[i];
}

__global__ void zero_f16_kernel(_Float16* dst, int n) {
  int i = blockIdx.x * 256 + threadIdx.x;
  if (i < n) dst[i] = (_Float16)0.f;
}

__global__ void copy_f32_kernel(const float* src, float* dst, int n) {
  int i = blockIdx.x * 256 + threadIdx.x;
  if (i < n) dst[i] = src[i];
}

// vt[b][c][k] = v16p[b][k][c]  (c in 0..KD-1, k in 0..TCP-1)
__global__ void transpose_v_kernel(const _Float16* __restrict__ v16p,
                                   _Float16* __restrict__ vt) {
  int idx = blockIdx.x * 256 + threadIdx.x;      // B*KD*TCP
  int k = idx % TCP_;
  int c = (idx / TCP_) % KD_;
  int b = idx / (TCP_ * KD_);
  vt[idx] = v16p[((size_t)b * TCP_ + k) * KD_ + c];
}

// stInj[b,d] = state[b] . Wsi[d] + bsi[d]
__global__ void state_proj_kernel(const float* state, const float* Wsi,
                                  const float* bsi, float* stInj) {
  int idx = blockIdx.x * 256 + threadIdx.x;      // B*D
  int b = idx >> 10, d = idx & 1023;
  const float* st = state + b * SDIM_;
  const float* w  = Wsi + (size_t)d * SDIM_;
  float s = bsi[d];
  for (int k = 0; k < SDIM_; ++k) s += st[k] * w[k];
  stInj[idx] = s;
}

// h_in rows (padded layout b*TCP+t): globals copied, chunk rows LN(x)*g+b + stInj
__global__ void prep_hin_kernel(const float* x, const float* gtok,
                                const float* lng, const float* lnb,
                                const float* stInj, _Float16* hin16, int chunkIdx) {
  int row = blockIdx.x;                          // 0..B*TC-1 (valid rows only)
  int b = row / TC_, t = row % TC_;
  int tid = threadIdx.x;                         // 256
  _Float16* hr = hin16 + ((size_t)b * TCP_ + t) * D_;
  if (t < MG_) {
    const float* src = gtok + (size_t)t * D_;
    for (int d = tid; d < D_; d += 256) hr[d] = (_Float16)src[d];
    return;                                      // block-uniform
  }
  int j = t - MG_;
  const float* xr = x + ((size_t)b * T_ + (size_t)chunkIdx * CHUNK_ + j) * D_;
  __shared__ float s1[256], s2[256];
  float vals[4]; float a = 0.f, q = 0.f;
#pragma unroll
  for (int i = 0; i < 4; ++i) {
    float v = xr[tid + 256 * i]; vals[i] = v; a += v; q += v * v;
  }
  s1[tid] = a; s2[tid] = q; __syncthreads();
  for (int off = 128; off > 0; off >>= 1) {
    if (tid < off) { s1[tid] += s1[tid + off]; s2[tid] += s2[tid + off]; }
    __syncthreads();
  }
  float m = s1[0] * (1.f / D_);
  float var = s2[0] * (1.f / D_) - m * m;
  float inv = rsqrtf(var + 1e-5f);
  const float* inj = stInj + (size_t)b * D_;
#pragma unroll
  for (int i = 0; i < 4; ++i) {
    int d = tid + 256 * i;
    hr[d] = (_Float16)((vals[i] - m) * inv * lng[d] + lnb[d] + inj[d]);
  }
}

// causal softmax row; writes all TCP cols (pad cols masked -> exact 0)
__global__ void softmax_kernel(const float* __restrict__ scores,
                               _Float16* __restrict__ attn) {
  int row = blockIdx.x, bh = blockIdx.y, tid = threadIdx.x;
  const float* sr = scores + ((size_t)bh * TCP_ + row) * TCP_;
  _Float16* ar    = attn   + ((size_t)bh * TCP_ + row) * TCP_;
  __shared__ float red[256];
  int t1 = tid + 256;
  float v0 = (tid <= row) ? sr[tid] * SCALE_ : -3.4e38f;
  float v1 = (t1  <= row) ? sr[t1]  * SCALE_ : -3.4e38f;
  red[tid] = fmaxf(v0, v1); __syncthreads();
  for (int off = 128; off > 0; off >>= 1) {
    if (tid < off) red[tid] = fmaxf(red[tid], red[tid + off]);
    __syncthreads();
  }
  float mx = red[0]; __syncthreads();
  float e0 = (tid <= row) ? expf(v0 - mx) : 0.f;
  float e1 = (t1  <= row) ? expf(v1 - mx) : 0.f;
  red[tid] = e0 + e1; __syncthreads();
  for (int off = 128; off > 0; off >>= 1) {
    if (tid < off) red[tid] += red[tid + off];
    __syncthreads();
  }
  float rinv = 1.f / red[0];
  ar[tid] = (_Float16)(e0 * rinv);
  if (t1 < TCP_) ar[t1] = (_Float16)(e1 * rinv);
}

// c = x_chunk + y(globals dropped, padded rows); cln16 = LN(c)*g+b
__global__ void residual_ln_kernel(const float* x, const float* y,
                                   const float* gf, const float* bf,
                                   float* c, _Float16* cln, int chunkIdx) {
  int row = blockIdx.x;                          // 0..B*CHUNK-1
  int b = row >> 8, j = row & 255;
  int tid = threadIdx.x;
  const float* xr = x + ((size_t)b * T_ + (size_t)chunkIdx * CHUNK_ + j) * D_;
  const float* yr = y + ((size_t)b * TCP_ + MG_ + j) * D_;
  float* cr = c + (size_t)row * D_;
  __shared__ float s1[256], s2[256];
  float vals[4]; float a = 0.f, q = 0.f;
#pragma unroll
  for (int i = 0; i < 4; ++i) {
    int d = tid + 256 * i;
    float v = xr[d] + yr[d];
    vals[i] = v; cr[d] = v; a += v; q += v * v;
  }
  s1[tid] = a; s2[tid] = q; __syncthreads();
  for (int off = 128; off > 0; off >>= 1) {
    if (tid < off) { s1[tid] += s1[tid + off]; s2[tid] += s2[tid + off]; }
    __syncthreads();
  }
  float m = s1[0] * (1.f / D_);
  float var = s2[0] * (1.f / D_) - m * m;
  float inv = rsqrtf(var + 1e-5f);
#pragma unroll
  for (int i = 0; i < 4; ++i) {
    int d = tid + 256 * i;
    cln[(size_t)row * D_ + d] = (_Float16)((vals[i] - m) * inv * gf[d] + bf[d]);
  }
}

// out = c + ff2 written straight into d_out at the chunk slot
__global__ void finish_kernel(const float* c, const float* ff2, float* out,
                              int chunkIdx) {
  int row = blockIdx.x; int b = row >> 8, j = row & 255;
  int tid = threadIdx.x;
  const float* fr = ff2 + (size_t)row * D_;
  const float* cr = c   + (size_t)row * D_;
  float* orow = out + ((size_t)b * T_ + (size_t)chunkIdx * CHUNK_ + j) * D_;
#pragma unroll
  for (int i = 0; i < 4; ++i) {
    int d = tid + 256 * i;
    orow[d] = cr[d] + fr[d];
  }
}

__global__ void chunk_mean_kernel(const float* out, float* mean, int chunkIdx) {
  int idx = blockIdx.x * 256 + threadIdx.x;      // B*D
  int b = idx >> 10, d = idx & 1023;
  const float* p = out + ((size_t)b * T_ + (size_t)chunkIdx * CHUNK_) * D_ + d;
  float s = 0.f;
  for (int j = 0; j < CHUNK_; ++j) s += p[(size_t)j * D_];
  mean[idx] = s * (1.f / CHUNK_);
}

__global__ void state_update_kernel(const float* mean, const float* Wsp,
                                    const float* bsp, float* state) {
  int idx = blockIdx.x * 256 + threadIdx.x;      // B*SDIM = 512
  if (idx >= B_ * SDIM_) return;
  int b = idx >> 7, s0 = idx & 127;
  const float* mn = mean + (size_t)b * D_;
  const float* w  = Wsp + (size_t)s0 * D_;
  float s = bsp[s0];
  for (int k = 0; k < D_; ++k) s += mn[k] * w[k];
  state[idx] = s;
}

// ---------------- host orchestration ----------------
extern "C" void kernel_launch(void* const* d_in, const int* in_sizes, int n_in,
                              void* d_out, int out_size, void* d_ws, size_t ws_size,
                              hipStream_t stream) {
  (void)in_sizes; (void)n_in; (void)out_size; (void)ws_size;
  const float* x    = (const float*)d_in[0];
  const float* st0  = (const float*)d_in[1];
  const float* Wq   = (const float*)d_in[2];
  const float* Wk   = (const float*)d_in[3];
  const float* Wv   = (const float*)d_in[4];
  const float* Wo   = (const float*)d_in[5];
  const float* bo   = (const float*)d_in[6];
  const float* lag  = (const float*)d_in[7];
  const float* lab  = (const float*)d_in[8];
  const float* lfg  = (const float*)d_in[9];
  const float* lfb  = (const float*)d_in[10];
  const float* W1   = (const float*)d_in[11];
  const float* b1   = (const float*)d_in[12];
  const float* W2   = (const float*)d_in[13];
  const float* b2   = (const float*)d_in[14];
  const float* Wsp  = (const float*)d_in[15];
  const float* bsp  = (const float*)d_in[16];
  const float* Wsi  = (const float*)d_in[17];
  const float* bsi  = (const float*)d_in[18];
  const float* gtok = (const float*)d_in[19];
  float* out = (float*)d_out;

  char* p = (char*)d_ws;
  auto alloc = [&](size_t bytes) -> void* {
    void* r = (void*)p; p += (bytes + 255) & ~(size_t)255; return r;
  };
  const size_t BC = (size_t)B_ * CHUNK_;
  _Float16* wq16   = (_Float16*)alloc((size_t)D_ * D_ * 2);
  _Float16* wk16   = (_Float16*)alloc((size_t)KD_ * D_ * 2);
  _Float16* wv16   = (_Float16*)alloc((size_t)KD_ * D_ * 2);
  _Float16* wo16   = (_Float16*)alloc((size_t)D_ * D_ * 2);
  _Float16* w1_16  = (_Float16*)alloc((size_t)4 * D_ * D_ * 2);
  _Float16* w2_16  = (_Float16*)alloc((size_t)4 * D_ * D_ * 2);
  float*    curst  = (float*)alloc((size_t)B_ * SDIM_ * 4);
  float*    stInj  = (float*)alloc((size_t)B_ * D_ * 4);
  float*    meanb  = (float*)alloc((size_t)B_ * D_ * 4);
  _Float16* hin16  = (_Float16*)alloc((size_t)BTCP_ * D_ * 2);
  _Float16* q16p   = (_Float16*)alloc((size_t)BTCP_ * D_ * 2);
  _Float16* k16p   = (_Float16*)alloc((size_t)BTCP_ * KD_ * 2);
  _Float16* v16p   = (_Float16*)alloc((size_t)BTCP_ * KD_ * 2);
  _Float16* vt16   = (_Float16*)alloc((size_t)B_ * KD_ * TCP_ * 2);
  float*    scores = (float*)alloc((size_t)BH_ * TCP_ * TCP_ * 4);
  _Float16* attn16 = (_Float16*)alloc((size_t)BH_ * TCP_ * TCP_ * 2);
  _Float16* o16p   = (_Float16*)alloc((size_t)BTCP_ * D_ * 2);
  float*    ybuf   = (float*)alloc((size_t)BTCP_ * D_ * 4);
  float*    cbuf   = (float*)alloc(BC * D_ * 4);
  _Float16* cln16  = (_Float16*)alloc(BC * D_ * 2);
  _Float16* ff1    = (_Float16*)alloc(BC * 4 * D_ * 2);
  float*    ff2    = (float*)alloc(BC * D_ * 4);

  auto cvt = [&](const float* src, _Float16* dst, int n) {
    f32_to_f16_kernel<<<dim3((n + 255) / 256), dim3(256), 0, stream>>>(src, dst, n);
  };
  cvt(Wq, wq16, D_ * D_);
  cvt(Wk, wk16, KD_ * D_);
  cvt(Wv, wv16, KD_ * D_);
  cvt(Wo, wo16, D_ * D_);
  cvt(W1, w1_16, 4 * D_ * D_);
  cvt(W2, w2_16, 4 * D_ * D_);
  copy_f32_kernel<<<dim3(2), dim3(256), 0, stream>>>(st0, curst, B_ * SDIM_);
  // zero padded buffers once: hin pad rows stay 0 => q/k/v pad rows exactly 0;
  // attn pad rows deterministic 0.
  {
    int nh = BTCP_ * D_;
    zero_f16_kernel<<<dim3((nh + 255) / 256), dim3(256), 0, stream>>>(hin16, nh);
    int na = BH_ * TCP_ * TCP_;
    zero_f16_kernel<<<dim3((na + 255) / 256), dim3(256), 0, stream>>>(attn16, na);
  }

  for (int ch = 0; ch < NCH_; ++ch) {
    state_proj_kernel<<<dim3((B_ * D_) / 256), dim3(256), 0, stream>>>(curst, Wsi, bsi, stInj);
    prep_hin_kernel<<<dim3(B_ * TC_), dim3(256), 0, stream>>>(x, gtok, lag, lab, stInj, hin16, ch);
    // QKV projections (padded M = 1152; pad rows are exact zeros)
    gemm_tpl<BTCP_, D_, D_, D_, D_, D_, false, 0, true, false>
        <<<dim3(D_ / 64, BTCP_ / 128), dim3(128), 0, stream>>>(hin16, wq16, q16p, nullptr);
    gemm_tpl<BTCP_, KD_, D_, D_, D_, KD_, false, 0, true, false>
        <<<dim3(KD_ / 64, BTCP_ / 128), dim3(128), 0, stream>>>(hin16, wk16, k16p, nullptr);
    gemm_tpl<BTCP_, KD_, D_, D_, D_, KD_, false, 0, true, false>
        <<<dim3(KD_ / 64, BTCP_ / 128), dim3(128), 0, stream>>>(hin16, wv16, v16p, nullptr);
    transpose_v_kernel<<<dim3((B_ * KD_ * TCP_) / 256), dim3(256), 0, stream>>>(v16p, vt16);
    // attention
    attn_scores_kernel<<<dim3(5, 3, BH_), dim3(128), 0, stream>>>(q16p, k16p, scores);
    softmax_kernel<<<dim3(TC_, BH_), dim3(256), 0, stream>>>(scores, attn16);
    attn_av_kernel<<<dim3(1, 3, BH_), dim3(128), 0, stream>>>(attn16, vt16, o16p);
    // output projection + bias
    gemm_tpl<BTCP_, D_, D_, D_, D_, D_, false, 0, false, true>
        <<<dim3(D_ / 64, BTCP_ / 128), dim3(128), 0, stream>>>(o16p, wo16, ybuf, bo);
    residual_ln_kernel<<<dim3(B_ * CHUNK_), dim3(256), 0, stream>>>(
        x, ybuf, lfg, lfb, cbuf, cln16, ch);
    // FFN (M = B*CHUNK = 1024)
    gemm_tpl<1024, 4 * D_, D_, D_, D_, 4 * D_, false, 1, true, true>
        <<<dim3((4 * D_) / 64, 1024 / 128), dim3(128), 0, stream>>>(cln16, w1_16, ff1, b1);
    gemm_tpl<1024, D_, 4 * D_, 4 * D_, 4 * D_, D_, false, 0, false, true>
        <<<dim3(D_ / 64, 1024 / 128), dim3(128), 0, stream>>>(ff1, w2_16, ff2, b2);
    finish_kernel<<<dim3(B_ * CHUNK_), dim3(256), 0, stream>>>(cbuf, ff2, out, ch);
    // state recurrence
    chunk_mean_kernel<<<dim3((B_ * D_) / 256), dim3(256), 0, stream>>>(out, meanb, ch);
    state_update_kernel<<<dim3(2), dim3(256), 0, stream>>>(meanb, Wsp, bsp, curst);
  }
  // final state appended after y_full
  copy_f32_kernel<<<dim3(2), dim3(256), 0, stream>>>(
      curst, out + (size_t)B_ * T_ * D_, B_ * SDIM_);
}
